// SelfAttention_v1_78176994722164
// MI455X (gfx1250) — matrix operand order repaired and checked by reading, not compiled, not run on hardware
//
#include <hip/hip_runtime.h>

// Self-attention, S=4096, D_IN=D_OUT=1024, fp32 in/out.
// Compute-bound (~95 GFLOP vs <200MB traffic @23.3TB/s) -> all GEMMs via
// v_wmma_f32_16x16x32_f16 (wave32, gfx1250). Tiles staged with
// global_load_async_to_lds_b128 into a 3-stage LDS ring; ASYNCcnt batch
// counting (s_wait_asynccnt 6) gives each transfer a full K-step of slack.

#define S_DIM 4096
#define D_DIM 1024

typedef __attribute__((ext_vector_type(8)))  _Float16 v8h;
typedef __attribute__((ext_vector_type(16))) _Float16 v16h;
typedef __attribute__((ext_vector_type(8)))  float    v8f;

union F16Frag { v16h v; struct { v8h lo, hi; } p; };

// CDNA5 async copy: 16B global -> LDS, tracked by ASYNCcnt.
__device__ __forceinline__ void async_ld16(uint32_t lds_byte, const void* g)
{
    asm volatile("global_load_async_to_lds_b128 %0, %1, off"
                 :: "v"(lds_byte), "v"((unsigned long long)(uintptr_t)g)
                 : "memory");
}
// Allow the newest 6-load batch to stay in flight; older batches retired.
__device__ __forceinline__ void wait_async_le6()
{
    asm volatile("s_wait_asynccnt 0x6" ::: "memory");
}
__device__ __forceinline__ void wait_async_all()
{
    asm volatile("s_wait_asynccnt 0x0" ::: "memory");
}

// ---------------------------------------------------------------------------
// NT GEMM: C[M,N] = alpha * A[M,K] * B[N,K]^T  (A,B f16 row-major along K)
//   MODE 0: f16 C[m*ldc+n]   MODE 1: f16 C[n*ldc+m]   MODE 2: f32 C[m*ldc+n]
// Block 256 thr = 8 wave32 in 4x2; block tile 256x128, BK=32.
// Wave tile 64x64 = 4x4 WMMA 16x16 (128 acc VGPRs/lane).
// 3-stage LDS ring: iter k issues stage k+2, computes stage k, waits
// asynccnt<=6 (stage k+1 landed, k+2 may fly), barrier publishes k+1.
// ---------------------------------------------------------------------------
template<int MODE>
__global__ __launch_bounds__(256)
void gemm_nt(const _Float16* __restrict__ A, const _Float16* __restrict__ B,
             void* __restrict__ Cp, int lda, int ldb, int ldc, int Kdim,
             float alpha)
{
    constexpr int BM = 256, BN = 128, BK = 32, LSTR = 40; // 80B row: 20-bank skew
    __shared__ _Float16 sA[3 * BM * LSTR];
    __shared__ _Float16 sB[3 * BN * LSTR];

    const int tid   = threadIdx.x;
    const int lane  = tid & 31;
    const int wid   = tid >> 5;       // 0..7
    const int waveM = wid >> 1;       // 0..3  (64-row slabs)
    const int waveN = wid & 1;        // 0..1  (64-col slabs)
    const int g     = lane >> 4;      // half-wave group (ISA K grouping)
    const int lm    = lane & 15;

    const int mBase = blockIdx.y * BM;
    const int nBase = blockIdx.x * BN;

    const uint32_t sA0 = (uint32_t)(uintptr_t)&sA[0];
    const uint32_t sB0 = (uint32_t)(uintptr_t)&sB[0];

    v8f acc[4][4] = {};

    // cooperative async-load mapping: 16B chunks, 4 chunks per 32-half row
    const int rA = tid >> 2;          // base row 0..63 (stride 64)
    const int q  = (tid & 3) * 8;     // half offset in row

    const int nsteps = Kdim / BK;

    auto load_stage = [&](int buf, int k0) {   // 6 async b128 per thread
        const uint32_t aB = sA0 + (uint32_t)(buf * BM * LSTR * 2);
        const uint32_t bB = sB0 + (uint32_t)(buf * BN * LSTR * 2);
        #pragma unroll
        for (int i = 0; i < 4; ++i) {                 // 256 rows of A
            const int r = rA + i * 64;
            async_ld16(aB + (uint32_t)(r * LSTR + q) * 2,
                       &A[(size_t)(mBase + r) * lda + k0 + q]);
        }
        #pragma unroll
        for (int i = 0; i < 2; ++i) {                 // 128 rows of B
            const int r = rA + i * 64;
            async_ld16(bB + (uint32_t)(r * LSTR + q) * 2,
                       &B[(size_t)(nBase + r) * ldb + k0 + q]);
        }
    };

    // prologue: stages 0 and 1 in flight; require stage 0 before compute
    load_stage(0, 0);
    if (nsteps > 1) load_stage(1, BK);
    wait_async_le6();
    __syncthreads();

    for (int step = 0; step < nsteps; ++step) {
        const int cur = step % 3;
        const bool issue = (step + 2) < nsteps;
        if (issue) load_stage((step + 2) % 3, (step + 2) * BK);

        const _Float16* a_s = &sA[cur * BM * LSTR];
        const _Float16* b_s = &sB[cur * BN * LSTR];

        // A frag (16x32): lane M=lm; K in [8g,8g+8) U [8g+16,8g+24)
        F16Frag a[4];
        #pragma unroll
        for (int i = 0; i < 4; ++i) {
            const int row = waveM * 64 + i * 16 + lm;
            a[i].p.lo = *(const v8h*)&a_s[row * LSTR + g * 8];
            a[i].p.hi = *(const v8h*)&a_s[row * LSTR + g * 8 + 16];
        }
        // B frag (32x16): lane N=lm; K in [8g,8g+8) U [8g+16,8g+24)
        F16Frag b[4];
        #pragma unroll
        for (int j = 0; j < 4; ++j) {
            const int row = waveN * 64 + j * 16 + lm;
            b[j].p.lo = *(const v8h*)&b_s[row * LSTR + g * 8];
            b[j].p.hi = *(const v8h*)&b_s[row * LSTR + g * 8 + 16];
        }

        #pragma unroll
        for (int i = 0; i < 4; ++i)
            #pragma unroll
            for (int j = 0; j < 4; ++j)
                acc[i][j] = __builtin_amdgcn_wmma_f32_16x16x32_f16(
                    false, a[i].v, false, b[j].v, (short)0, acc[i][j],
                    false, false);

        // retire stage step+1's writes (newest batch may remain in flight)
        if (issue) wait_async_le6();
        else       wait_async_all();
        __syncthreads();
    }

    // Epilogue. C/D layout: n = lane%16, m = vgpr + 8*(lane/16).
    const int mW = mBase + waveM * 64 + g * 8;
    const int nW = nBase + waveN * 64 + lm;
    #pragma unroll
    for (int i = 0; i < 4; ++i) {
        #pragma unroll
        for (int j = 0; j < 4; ++j) {
            const int n = nW + j * 16;
            #pragma unroll
            for (int r = 0; r < 8; ++r) {
                const int m = mW + i * 16 + r;
                const float v = acc[i][j][r] * alpha;
                if (MODE == 0)
                    ((_Float16*)Cp)[(size_t)m * ldc + n] = (_Float16)v;
                else if (MODE == 1)
                    ((_Float16*)Cp)[(size_t)n * ldc + m] = (_Float16)v;
                else
                    ((float*)Cp)[(size_t)m * ldc + n] = v;
            }
        }
    }
}

// ---------------------------------------------------------------------------
__global__ void cvt_f32_f16(const float* __restrict__ in,
                            _Float16* __restrict__ out, int n)
{
    for (int i = blockIdx.x * blockDim.x + threadIdx.x; i < n;
         i += gridDim.x * blockDim.x)
        out[i] = (_Float16)in[i];
}

// fp32 [rows x cols] -> f16 transposed [cols x rows], LDS-tiled (32x32)
__global__ void transpose_cvt(const float* __restrict__ in,
                              _Float16* __restrict__ out, int rows, int cols)
{
    __shared__ float tile[32][33];
    const int bx = blockIdx.x * 32, by = blockIdx.y * 32;
    const int tx = threadIdx.x, ty = threadIdx.y;
    #pragma unroll
    for (int r = ty; r < 32; r += 8)
        tile[r][tx] = in[(size_t)(by + r) * cols + bx + tx];
    __syncthreads();
    #pragma unroll
    for (int r = ty; r < 32; r += 8)
        out[(size_t)(bx + r) * rows + by + tx] = (_Float16)tile[tx][r];
}

// In-place row softmax on f16 scores (f32 math). One block per row.
__global__ __launch_bounds__(256)
void softmax_inplace(_Float16* __restrict__ s, int ncols)
{
    __shared__ float red[256];
    const int tid = threadIdx.x;
    _Float16* p = s + (size_t)blockIdx.x * ncols;

    float mx = -1e30f;
    for (int i = tid; i < ncols; i += 256) mx = fmaxf(mx, (float)p[i]);
    red[tid] = mx; __syncthreads();
    for (int o = 128; o > 0; o >>= 1) {
        if (tid < o) red[tid] = fmaxf(red[tid], red[tid + o]);
        __syncthreads();
    }
    mx = red[0]; __syncthreads();

    float sum = 0.f;
    for (int i = tid; i < ncols; i += 256) sum += __expf((float)p[i] - mx);
    red[tid] = sum; __syncthreads();
    for (int o = 128; o > 0; o >>= 1) {
        if (tid < o) red[tid] += red[tid + o];
        __syncthreads();
    }
    const float inv = 1.0f / red[0];

    for (int i = tid; i < ncols; i += 256)
        p[i] = (_Float16)(__expf((float)p[i] - mx) * inv);
}

// ---------------------------------------------------------------------------
extern "C" void kernel_launch(void* const* d_in, const int* in_sizes, int n_in,
                              void* d_out, int out_size, void* d_ws, size_t ws_size,
                              hipStream_t stream)
{
    const float* x  = (const float*)d_in[0];
    const float* Wq = (const float*)d_in[1];
    const float* Wk = (const float*)d_in[2];
    const float* Wv = (const float*)d_in[3];

    // workspace carve-out (~70 MB), 256B aligned
    char* w = (char*)d_ws;
    auto take = [&](size_t bytes) -> char* {
        char* p = w;
        w += (bytes + 255) & ~(size_t)255;
        return p;
    };
    _Float16* xb  = (_Float16*)take((size_t)S_DIM * D_DIM * 2);
    _Float16* Qm  = (_Float16*)take((size_t)S_DIM * D_DIM * 2);
    _Float16* Km  = (_Float16*)take((size_t)S_DIM * D_DIM * 2);
    _Float16* Vt  = (_Float16*)take((size_t)S_DIM * D_DIM * 2);  // V^T [D x S]
    _Float16* Wqt = (_Float16*)take((size_t)D_DIM * D_DIM * 2);
    _Float16* Wkt = (_Float16*)take((size_t)D_DIM * D_DIM * 2);
    _Float16* Wvt = (_Float16*)take((size_t)D_DIM * D_DIM * 2);
    _Float16* Sc  = (_Float16*)take((size_t)S_DIM * S_DIM * 2);  // scores / P

    // 1) precision convert + weight transposes (NT GEMM form)
    cvt_f32_f16<<<1024, 256, 0, stream>>>(x, xb, S_DIM * D_DIM);
    {
        dim3 tb(32, 8), tg(D_DIM / 32, D_DIM / 32);
        transpose_cvt<<<tg, tb, 0, stream>>>(Wq, Wqt, D_DIM, D_DIM);
        transpose_cvt<<<tg, tb, 0, stream>>>(Wk, Wkt, D_DIM, D_DIM);
        transpose_cvt<<<tg, tb, 0, stream>>>(Wv, Wvt, D_DIM, D_DIM);
    }

    const dim3 blk(256);

    // 2) QKV projections: Q pre-scaled by 1/sqrt(1024); V stored transposed.
    const dim3 gQKV(D_DIM / 128, S_DIM / 256);   // (8, 16)
    gemm_nt<0><<<gQKV, blk, 0, stream>>>(xb, Wqt, Qm, D_DIM, D_DIM, D_DIM, D_DIM, 0.03125f);
    gemm_nt<0><<<gQKV, blk, 0, stream>>>(xb, Wkt, Km, D_DIM, D_DIM, D_DIM, D_DIM, 1.0f);
    gemm_nt<1><<<gQKV, blk, 0, stream>>>(xb, Wvt, Vt, D_DIM, D_DIM, S_DIM, D_DIM, 1.0f);

    // 3) scores = (Q/sqrt(d)) K^T : [4096x4096] f16
    const dim3 gS(S_DIM / 128, S_DIM / 256);     // (32, 16)
    gemm_nt<0><<<gS, blk, 0, stream>>>(Qm, Km, Sc, D_DIM, D_DIM, S_DIM, D_DIM, 1.0f);

    // 4) row softmax in place
    softmax_inplace<<<S_DIM, 256, 0, stream>>>(Sc, S_DIM);

    // 5) out(f32) = P V, using V^T as the NT B operand
    const dim3 gO(D_DIM / 128, S_DIM / 256);     // (8, 16)
    gemm_nt<2><<<gO, blk, 0, stream>>>(Sc, Vt, d_out, S_DIM, S_DIM, D_DIM, S_DIM, 1.0f);
}
